// RawWaveformTokenizer_35296041239261
// MI455X (gfx1250) — matrix operand order repaired and checked by reading, not compiled, not run on hardware
//
#include <hip/hip_runtime.h>
#include <hip/hip_bf16.h>

typedef __attribute__((ext_vector_type(16))) _Float16 v16h;
typedef __attribute__((ext_vector_type(8)))  float    v8f;

// ---------------- workspace layout (bytes, 256-aligned) ----------------
#define ACT_ELEMS   40960000ull            // every layer output: 16*C*L == 40.96M
#define OFF_ACTA    0ull
#define OFF_ACTB    163840000ull
#define OFF_CBF16   327680000ull           // 8*1024*1024 f16
#define OFF_PWF16   344457216ull           // 1024*1024 f16
#define OFF_WF2     346554368ull           // 64*160 f16 (A-frag swizzled)
#define OFF_WF3     346574848ull           // 128*320
#define OFF_WF4     346656768ull           // 256*384
#define OFF_WF5     346853376ull           // 512*768
#define OFF_WF6     347639808ull           // 1024*1536
#define OFF_CNORM   350785536ull           // 8192 f32
#define OFF_STATS   350818304ull           // 256 f32 (mean,rstd per b*8+g)
#define OFF_LPART   350819328ull           // 2500 f32 loss partials

#define WMMA_F16(a, b, c) \
    __builtin_amdgcn_wmma_f32_16x16x32_f16(false, (a), false, (b), (short)0, (c), false, false)

// ---------------- small utility kernels ----------------
__global__ void cvt_f32_f16_kernel(const float* __restrict__ src,
                                   _Float16* __restrict__ dst, int n) {
    int i = blockIdx.x * 256 + threadIdx.x;
    if (i < n) dst[i] = (_Float16)src[i];
}

__global__ void zero_kernel(float* __restrict__ p, int n) {
    int i = blockIdx.x * 256 + threadIdx.x;
    if (i < n) p[i] = 0.f;
}

// Swizzle conv weights [co][kk] -> A-fragment order:
// block (co>>4), step ks, lane (co&15)+16*h, element e; kk=ks*32+kkl,
// h=(kkl>>3)&1, e=(kkl&7)+(kkl&16?8:0)
__global__ void wswz_kernel(const float* __restrict__ src, _Float16* __restrict__ dst,
                            int Kdim) {
    int kkf = blockIdx.x * 256 + threadIdx.x;
    int co  = blockIdx.y;
    if (kkf >= Kdim) return;
    int ks = kkf >> 5, kkl = kkf & 31;
    int h = (kkl >> 3) & 1;
    int e = (kkl & 7) + ((kkl & 16) ? 8 : 0);
    int lane = (co & 15) + 16 * h;
    size_t dstIdx = ((size_t)(co >> 4) * (Kdim >> 5) + ks) * 512 + lane * 16 + e;
    dst[dstIdx] = (_Float16)src[(size_t)co * Kdim + kkf];
}

// per-code squared norm of codebooks (8*1024 codes, H=1024)
__global__ void cnorm_kernel(const float* __restrict__ cb, float* __restrict__ cn) {
    __shared__ float s[128];
    int code = blockIdx.x;
    const float* row = cb + (size_t)code * 1024;
    float v = 0.f;
    for (int i = threadIdx.x; i < 1024; i += 128) { float x = row[i]; v += x * x; }
    s[threadIdx.x] = v; __syncthreads();
    for (int st = 64; st; st >>= 1) { if (threadIdx.x < st) s[threadIdx.x] += s[threadIdx.x + st]; __syncthreads(); }
    if (!threadIdx.x) cn[code] = s[0];
}

// ---------------- layer 1: C_in=1, k=7, stride 2, direct f32 ----------------
__global__ void conv1_kernel(const float* __restrict__ wav, const float* __restrict__ w,
                             const float* __restrict__ bias, float* __restrict__ y) {
    int t  = blockIdx.x * 256 + threadIdx.x;
    int co = blockIdx.y, b = blockIdx.z;
    if (t >= 80000) return;
    float acc = bias[co];
    const float* xb = wav + (size_t)b * 160000;
    #pragma unroll
    for (int j = 0; j < 7; ++j) {
        int tin = 2 * t + j - 3;
        if (tin >= 0 && tin < 160000) acc += w[co * 7 + j] * xb[tin];
    }
    y[((size_t)(b * 32 + co)) * 80000 + t] = acc;
}

// ---------------- layers 2-6: implicit GEMM via WMMA f16 ----------------
// B panel staged in LDS pre-swizzled to fragment order: [ks][lane][e]
template <int KSZ>
__global__ __launch_bounds__(256) void conv_wmma_kernel(
    const float* __restrict__ x, float* __restrict__ y,
    const _Float16* __restrict__ wfs, const float* __restrict__ bias,
    int Cin, int Cout, int Lin, int Lout, int Kdim)
{
    __shared__ _Float16 Bs[1536 * 16];
    const int tid = threadIdx.x;
    const int wave = tid >> 5, lane = tid & 31;
    const int h = lane >> 4, nloc = lane & 15;
    const int t0 = blockIdx.x * 16;
    const int b = blockIdx.z;
    const int pad = KSZ >> 1;

    for (int i = tid; i < Kdim * 16; i += 256) {
        int kkf = i >> 4, n = i & 15;
        int ci = kkf / KSZ, j = kkf - ci * KSZ;     // KSZ is constexpr -> no div
        int tin = 2 * (t0 + n) + j - pad;
        float v = 0.f;
        if (tin >= 0 && tin < Lin) v = x[((size_t)(b * Cin + ci)) * Lin + tin];
        int ks = kkf >> 5, kkl = kkf & 31;
        int hb = (kkl >> 3) & 1, e = (kkl & 7) + ((kkl & 16) ? 8 : 0);            // B-fragment order
        Bs[(ks * 32 + n + 16 * hb) * 16 + e] = (_Float16)v;
    }
    __syncthreads();

    const int co_tile = blockIdx.y * 128 + wave * 16;
    if (co_tile >= Cout) return;     // wave-uniform

    const int nsteps = Kdim >> 5;
    const _Float16* wbase = wfs + (size_t)(co_tile >> 4) * nsteps * 512;
    v8f acc = {};
    for (int ks = 0; ks < nsteps; ++ks) {
        v16h a  = *(const v16h*)(wbase + ks * 512 + lane * 16);
        v16h bb = *(const v16h*)(Bs + (ks * 32 + lane) * 16);
        acc = WMMA_F16(a, bb, acc);
    }
    int t = t0 + nloc;
    if (t < Lout) {
        #pragma unroll
        for (int r = 0; r < 8; ++r) {
            int co = co_tile + r + 8 * h;
            y[((size_t)(b * Cout + co)) * Lout + t] = acc[r] + bias[co];
        }
    }
}

// ---------------- GroupNorm (8 groups) ----------------
__global__ void gn_stats_kernel(const float* __restrict__ x, float* __restrict__ stats,
                                int C, int L) {
    __shared__ float sa[256], sb[256];
    int bg = blockIdx.x;              // b*8+g
    int b = bg >> 3, g = bg & 7;
    int cpg = C >> 3;
    size_t base = ((size_t)b * C + (size_t)g * cpg) * (size_t)L;
    size_t n = (size_t)cpg * L;
    float s = 0.f, s2 = 0.f;
    for (size_t i = threadIdx.x; i < n; i += 256) { float v = x[base + i]; s += v; s2 += v * v; }
    sa[threadIdx.x] = s; sb[threadIdx.x] = s2; __syncthreads();
    for (int st = 128; st; st >>= 1) {
        if (threadIdx.x < st) { sa[threadIdx.x] += sa[threadIdx.x + st]; sb[threadIdx.x] += sb[threadIdx.x + st]; }
        __syncthreads();
    }
    if (!threadIdx.x) {
        float m = sa[0] / (float)n;
        float var = sb[0] / (float)n - m * m;
        stats[2 * bg] = m;
        stats[2 * bg + 1] = rsqrtf(var + 1e-5f);
    }
}

// grid: (ceil(L/256), C, 16); group = c >> cpgShift
__global__ void gn_apply_kernel(const float* __restrict__ x, float* __restrict__ y,
                                const float* __restrict__ stats,
                                const float* __restrict__ gam, const float* __restrict__ bet,
                                int C, int L, int cpgShift, int transpose) {
    int t = blockIdx.x * 256 + threadIdx.x;
    int c = blockIdx.y, b = blockIdx.z;
    if (t >= L) return;
    int bg = b * 8 + (c >> cpgShift);
    size_t idx = ((size_t)(b * C + c)) * L + t;
    float v = (x[idx] - stats[2 * bg]) * stats[2 * bg + 1] * gam[c] + bet[c];
    v = v / (1.f + __expf(-v));                          // SiLU
    if (transpose) y[((size_t)b * L + t) * C + c] = v;   // -> [b*T + t][H]
    else           y[idx] = v;
}

// stage 16 rows x 1024 f32 -> LDS f16 in A-fragment order, float4 reads
__device__ __forceinline__ void stage_rows_swz(const float* __restrict__ src,
                                               size_t row0, _Float16* __restrict__ Rs,
                                               int tid) {
    for (int i4 = tid; i4 < 4096; i4 += 256) {
        int m  = i4 >> 8;
        int k4 = (i4 & 255) << 2;
        const float4 v = *(const float4*)(src + (row0 + m) * 1024 + k4);
        int ks = k4 >> 5, kkl = k4 & 31;
        int h = kkl >> 4;
        int e0 = kkl & 15;
        _Float16* d = Rs + (ks * 32 + m + 16 * h) * 16 + e0;
        d[0] = (_Float16)v.x; d[1] = (_Float16)v.y;
        d[2] = (_Float16)v.z; d[3] = (_Float16)v.w;
    }
}

// ---------------- RVQ stage: fused GEMM-argmin + residual update ----------------
// 4 N-tiles per K-loop: 1 LDS A-frag feeds 4 independent WMMAs.
__global__ __launch_bounds__(256) void rvq_stage_kernel(
    float* __restrict__ residual, float* __restrict__ quant,
    const _Float16* __restrict__ cbf, const float* __restrict__ cb32,
    const float* __restrict__ cnorm, float* __restrict__ losspart, int stage)
{
    __shared__ _Float16 Rs[16 * 1024];
    __shared__ float sval[128];
    __shared__ int   sidx[128];
    __shared__ int   sfin[16];
    __shared__ float sred[256];

    const int tid = threadIdx.x, wave = tid >> 5, lane = tid & 31;
    const int h = lane >> 4, nloc = lane & 15;
    const size_t row0 = (size_t)blockIdx.x * 16;

    stage_rows_swz(residual, row0, Rs, tid);
    __syncthreads();

    const _Float16* cbs = cbf + (size_t)stage * 1024 * 1024;
    float best[8]; int bidx[8];
    #pragma unroll
    for (int r = 0; r < 8; ++r) { best[r] = 3.4e38f; bidx[r] = 0; }

    for (int g = 0; g < 2; ++g) {                    // 2 groups x 4 code-tiles
        int code0 = (wave * 8 + g * 4) * 16 + nloc;  // tiles: code0 + 16*q
        const _Float16* bcol = cbs + (size_t)code0 * 1024 + 16 * h;
        v8f acc[4] = {{}, {}, {}, {}};
        #pragma unroll 2
        for (int ks = 0; ks < 32; ++ks) {
            v16h a  = *(const v16h*)(Rs + (ks * 32 + lane) * 16);
            v16h b0 = *(const v16h*)(bcol + ks * 32);
            v16h b1 = *(const v16h*)(bcol + 16 * 1024 + ks * 32);
            v16h b2 = *(const v16h*)(bcol + 32 * 1024 + ks * 32);
            v16h b3 = *(const v16h*)(bcol + 48 * 1024 + ks * 32);
            acc[0] = WMMA_F16(a, b0, acc[0]);
            acc[1] = WMMA_F16(a, b1, acc[1]);
            acc[2] = WMMA_F16(a, b2, acc[2]);
            acc[3] = WMMA_F16(a, b3, acc[3]);
        }
        #pragma unroll
        for (int q = 0; q < 4; ++q) {                // ascending code order
            int code = code0 + 16 * q;
            float cn = cnorm[stage * 1024 + code];
            #pragma unroll
            for (int r = 0; r < 8; ++r) {
                float d = cn - 2.f * acc[q][r];
                if (d < best[r] || (d == best[r] && code < bidx[r])) { best[r] = d; bidx[r] = code; }
            }
        }
    }
    // reduce over the 16 lanes holding each row (argmin, lowest index on tie)
    #pragma unroll
    for (int r = 0; r < 8; ++r) {
        for (int mask = 8; mask; mask >>= 1) {
            float ov = __shfl_xor(best[r], mask, 32);
            int   oi = __shfl_xor(bidx[r], mask, 32);
            if (ov < best[r] || (ov == best[r] && oi < bidx[r])) { best[r] = ov; bidx[r] = oi; }
        }
    }
    if (nloc == 0) {
        #pragma unroll
        for (int r = 0; r < 8; ++r) {
            sval[wave * 16 + 8 * h + r] = best[r];
            sidx[wave * 16 + 8 * h + r] = bidx[r];
        }
    }
    __syncthreads();
    if (tid < 16) {    // reduce the 8 wave candidates per row (waves scan ascending code ranges)
        float bv = sval[tid]; int bi = sidx[tid];
        for (int w = 1; w < 8; ++w) {
            float v = sval[w * 16 + tid]; int ii = sidx[w * 16 + tid];
            if (v < bv || (v == bv && ii < bi)) { bv = v; bi = ii; }
        }
        sfin[tid] = bi;
    }
    __syncthreads();

    // update: quant += qs; loss += (qs - residual)^2; residual -= qs (float4)
    float lsum = 0.f;
    const float* cq = cb32 + (size_t)stage * 1024 * 1024;
    for (int i4 = tid; i4 < 4096; i4 += 256) {
        int m  = i4 >> 8;
        int k4 = (i4 & 255) << 2;
        size_t g = (row0 + m) * 1024 + k4;
        const float4 q4 = *(const float4*)(cq + (size_t)sfin[m] * 1024 + k4);
        float4 r4 = *(const float4*)(residual + g);
        float4 t4 = *(const float4*)(quant + g);
        t4.x += q4.x; t4.y += q4.y; t4.z += q4.z; t4.w += q4.w;
        float dx = q4.x - r4.x, dy = q4.y - r4.y, dz = q4.z - r4.z, dw = q4.w - r4.w;
        lsum += dx * dx + dy * dy + dz * dz + dw * dw;
        r4.x -= q4.x; r4.y -= q4.y; r4.z -= q4.z; r4.w -= q4.w;
        *(float4*)(quant + g) = t4;
        *(float4*)(residual + g) = r4;
    }
    sred[tid] = lsum; __syncthreads();
    for (int st = 128; st; st >>= 1) { if (tid < st) sred[tid] += sred[tid + st]; __syncthreads(); }
    if (!tid) losspart[blockIdx.x] += sred[0];   // one block per slot -> deterministic
}

// ---------------- output projection: out = quant @ W^T + b ----------------
__global__ __launch_bounds__(256) void proj_kernel(
    const float* __restrict__ quant, const _Float16* __restrict__ pw,
    const float* __restrict__ pb, float* __restrict__ out)
{
    __shared__ _Float16 Qs[16 * 1024];
    const int tid = threadIdx.x, wave = tid >> 5, lane = tid & 31;
    const int h = lane >> 4, nloc = lane & 15;
    const size_t row0 = (size_t)blockIdx.x * 16;

    stage_rows_swz(quant, row0, Qs, tid);
    __syncthreads();

    for (int g = 0; g < 2; ++g) {                    // 2 groups x 4 col-tiles
        int col0 = (wave * 8 + g * 4) * 16 + nloc;
        const _Float16* bcol = pw + (size_t)col0 * 1024 + 16 * h;
        v8f acc[4] = {{}, {}, {}, {}};
        #pragma unroll 2
        for (int ks = 0; ks < 32; ++ks) {
            v16h a  = *(const v16h*)(Qs + (ks * 32 + lane) * 16);
            v16h b0 = *(const v16h*)(bcol + ks * 32);
            v16h b1 = *(const v16h*)(bcol + 16 * 1024 + ks * 32);
            v16h b2 = *(const v16h*)(bcol + 32 * 1024 + ks * 32);
            v16h b3 = *(const v16h*)(bcol + 48 * 1024 + ks * 32);
            acc[0] = WMMA_F16(a, b0, acc[0]);
            acc[1] = WMMA_F16(a, b1, acc[1]);
            acc[2] = WMMA_F16(a, b2, acc[2]);
            acc[3] = WMMA_F16(a, b3, acc[3]);
        }
        #pragma unroll
        for (int q = 0; q < 4; ++q) {
            int col = col0 + 16 * q;
            float bc = pb[col];
            #pragma unroll
            for (int r = 0; r < 8; ++r)
                out[(row0 + r + 8 * h) * 1024 + col] = acc[q][r] + bc;
        }
    }
}

__global__ void loss_final_kernel(const float* __restrict__ lp, float* __restrict__ out) {
    __shared__ float s[256];
    float v = 0.f;
    for (int i = threadIdx.x; i < 2500; i += 256) v += lp[i];
    s[threadIdx.x] = v; __syncthreads();
    for (int st = 128; st; st >>= 1) { if (threadIdx.x < st) s[threadIdx.x] += s[threadIdx.x + st]; __syncthreads(); }
    if (!threadIdx.x) out[40960000] = 0.25f * s[0] / 40960000.f;
}

// ---------------- host orchestration ----------------
extern "C" void kernel_launch(void* const* d_in, const int* in_sizes, int n_in,
                              void* d_out, int out_size, void* d_ws, size_t ws_size,
                              hipStream_t stream) {
    (void)in_sizes; (void)n_in; (void)out_size; (void)ws_size;
    const float* wav = (const float*)d_in[0];
    const float* cw[6]; const float* cb[6]; const float* gg[6]; const float* gb[6];
    for (int i = 0; i < 6; ++i) {
        cw[i] = (const float*)d_in[1 + i];
        cb[i] = (const float*)d_in[7 + i];
        gg[i] = (const float*)d_in[13 + i];
        gb[i] = (const float*)d_in[19 + i];
    }
    const float* codebooks = (const float*)d_in[25];
    const float* proj_w    = (const float*)d_in[26];
    const float* proj_b    = (const float*)d_in[27];
    float* out = (float*)d_out;

    char* ws = (char*)d_ws;
    float*    actA   = (float*)(ws + OFF_ACTA);
    float*    actB   = (float*)(ws + OFF_ACTB);
    _Float16* cbf16  = (_Float16*)(ws + OFF_CBF16);
    _Float16* pwf16  = (_Float16*)(ws + OFF_PWF16);
    _Float16* wf[7]  = {nullptr, nullptr,
                        (_Float16*)(ws + OFF_WF2), (_Float16*)(ws + OFF_WF3),
                        (_Float16*)(ws + OFF_WF4), (_Float16*)(ws + OFF_WF5),
                        (_Float16*)(ws + OFF_WF6)};
    float* cnorm = (float*)(ws + OFF_CNORM);
    float* stats = (float*)(ws + OFF_STATS);
    float* lpart = (float*)(ws + OFF_LPART);

    const int GRID_E = (int)((ACT_ELEMS + 255) / 256);   // 160000

    // weight swizzle + codebook/proj conversions to f16
    struct { int Cin, Cout, k, Lin, Lout, Kdim, shf; } L[7] = {
        {}, {1, 32, 7, 160000, 80000, 7, 2},
        {32, 64, 5, 80000, 40000, 160, 3}, {64, 128, 5, 40000, 20000, 320, 4},
        {128, 256, 3, 20000, 10000, 384, 5}, {256, 512, 3, 10000, 5000, 768, 6},
        {512, 1024, 3, 5000, 2500, 1536, 7}};
    for (int l = 2; l <= 6; ++l)
        wswz_kernel<<<dim3((L[l].Kdim + 255) / 256, L[l].Cout), 256, 0, stream>>>(
            cw[l - 1], wf[l], L[l].Kdim);
    cvt_f32_f16_kernel<<<(8388608 + 255) / 256, 256, 0, stream>>>(codebooks, cbf16, 8388608);
    cvt_f32_f16_kernel<<<(1048576 + 255) / 256, 256, 0, stream>>>(proj_w, pwf16, 1048576);
    cnorm_kernel<<<8192, 128, 0, stream>>>(codebooks, cnorm);
    zero_kernel<<<(2500 + 255) / 256, 256, 0, stream>>>(lpart, 2500);

    // ---- encoder ----
    conv1_kernel<<<dim3(313, 32, 16), 256, 0, stream>>>(wav, cw[0], cb[0], actA);
    gn_stats_kernel<<<128, 256, 0, stream>>>(actA, stats, 32, 80000);
    gn_apply_kernel<<<dim3(313, 32, 16), 256, 0, stream>>>(actA, actA, stats, gg[0], gb[0],
                                                           32, 80000, 2, 0);

    float* bufs[2] = {actA, actB};
    for (int l = 2; l <= 6; ++l) {
        float* in   = bufs[l & 1];          // l=2: actA, l=3: actB, ...
        float* outp = bufs[(l + 1) & 1];
        dim3 grid((L[l].Lout + 15) / 16, (L[l].Cout + 127) / 128, 16);
        if (L[l].k == 5)
            conv_wmma_kernel<5><<<grid, 256, 0, stream>>>(in, outp, wf[l], cb[l - 1],
                L[l].Cin, L[l].Cout, L[l].Lin, L[l].Lout, L[l].Kdim);
        else
            conv_wmma_kernel<3><<<grid, 256, 0, stream>>>(in, outp, wf[l], cb[l - 1],
                L[l].Cin, L[l].Cout, L[l].Lin, L[l].Lout, L[l].Kdim);
        gn_stats_kernel<<<128, 256, 0, stream>>>(outp, stats, L[l].Cout, L[l].Lout);
        dim3 egrid((L[l].Lout + 255) / 256, L[l].Cout, 16);
        if (l < 6) {
            gn_apply_kernel<<<egrid, 256, 0, stream>>>(outp, outp, stats, gg[l - 1], gb[l - 1],
                                                       L[l].Cout, L[l].Lout, L[l].shf, 0);
        } else {
            // layer 6: apply + transpose [b][1024][2500] -> [(b*2500+t)][1024] into actA
            gn_apply_kernel<<<egrid, 256, 0, stream>>>(outp, actA, stats, gg[5], gb[5],
                                                       1024, 2500, 7, 1);
        }
    }
    // residual = actA (features), quant = actB (zeroed)
    zero_kernel<<<GRID_E, 256, 0, stream>>>(actB, (int)ACT_ELEMS);

    // ---- RVQ: 8 stages ----
    for (int s = 0; s < 8; ++s)
        rvq_stage_kernel<<<2500, 256, 0, stream>>>(actA, actB, cbf16, codebooks, cnorm, lpart, s);

    // ---- projection + loss ----
    proj_kernel<<<2500, 256, 0, stream>>>(actB, pwf16, proj_b, out);
    loss_final_kernel<<<1, 256, 0, stream>>>(lpart, out);
}
